// BotDCGCGraphAutoEncoder_88648124989885
// MI455X (gfx1250) — matrix operand order repaired and checked by reading, not compiled, not run on hardware
//
#include <hip/hip_runtime.h>
#include <hip/hip_bf16.h>

#define NN   20000
#define EM   800000
#define EPOS 160000
#define ENEG 320000
#define D0   512
#define D1   256
#define D2   128

typedef __bf16 bf16_t;
typedef bf16_t v16bf __attribute__((ext_vector_type(16)));
typedef float  v8f   __attribute__((ext_vector_type(8)));
typedef unsigned int v8u __attribute__((ext_vector_type(8)));

// ---------- helpers ----------

__device__ __forceinline__ unsigned int f2bf_pack(float lo, float hi) {
  unsigned int ul = __builtin_bit_cast(unsigned int, lo);
  ul = (ul + 0x7FFFu + ((ul >> 16) & 1u)) >> 16;
  unsigned int uh = __builtin_bit_cast(unsigned int, hi);
  uh = (uh + 0x7FFFu + ((uh >> 16) & 1u)) >> 16;
  return (ul & 0xFFFFu) | (uh << 16);
}

__device__ __forceinline__ float wave_sum(float v) {
#pragma unroll
  for (int off = 16; off > 0; off >>= 1) v += __shfl_xor(v, off, 32);
  return v;
}

// ---------- fragment packing ----------
// A fragment (16x32 bf16, ISA layout): lane l holds row M=l&15;
// VGPR v holds K = 16*(v>>2) + 8*(l>>4) + 2*(v&3) + {0,1}
template<int K>
__global__ __launch_bounds__(256)
void pack_a(const float* __restrict__ X, unsigned int* __restrict__ P, int M) {
  constexpr int NK = K / 32;
  int wid  = (int)((blockIdx.x * blockDim.x + threadIdx.x) >> 5);
  int lane = threadIdx.x & 31;
  if (wid >= (M >> 4) * NK) return;
  int tm = wid / NK, kk = wid % NK;
  int row = (tm << 4) + (lane & 15);
  int hi  = lane >> 4;
  const float* xr = X + (size_t)row * K + ((size_t)kk << 5);
  unsigned int* dst = P + ((size_t)wid * 32 + lane) * 8;
#pragma unroll
  for (int v = 0; v < 8; ++v) {
    int ka = ((v >> 2) << 4) + (hi << 3) + ((v & 3) << 1);
    float2 p = *(const float2*)(xr + ka);
    dst[v] = f2bf_pack(p.x, p.y);
  }
}

// B fragment (32x16 bf16, ISA layout): lane l holds col N=l&15;
// VGPR v holds K = 16*(v>>2) + 8*(l>>4) + 2*(v&3) + {0,1}
template<int K, int N>
__global__ __launch_bounds__(256)
void pack_b(const float* __restrict__ W, unsigned int* __restrict__ P) {
  constexpr int NK = K / 32;
  constexpr int TN = N / 16;
  int wid  = (int)((blockIdx.x * blockDim.x + threadIdx.x) >> 5);
  int lane = threadIdx.x & 31;
  if (wid >= TN * NK) return;
  int tn = wid / NK, kk = wid % NK;
  int ncol = (tn << 4) + (lane & 15);
  int hi   = lane >> 4;
  unsigned int* dst = P + ((size_t)wid * 32 + lane) * 8;
#pragma unroll
  for (int v = 0; v < 8; ++v) {
    int kb = (kk << 5) + ((v >> 2) << 4) + (hi << 3) + ((v & 3) << 1);
    dst[v] = f2bf_pack(W[(size_t)kb * N + ncol], W[(size_t)(kb + 1) * N + ncol]);
  }
}

// ---------- dual GEMM: C = A@B, C2 = A@B2 ----------
// Each wave computes a 16x(16*TW) strip for BOTH weight matrices.
// Per K-step: stage 1 A + 2*TW B fragments into distinct registers first,
// then issue 2*TW WMMAs -- lets the scheduler clause the loads, wait
// progressively, and prefetch the next K-step under the matrix ops.
template<int K, int N>
__global__ __launch_bounds__(256)
void gemm_dual_wmma(const unsigned int* __restrict__ Ap,
                    const unsigned int* __restrict__ Bp,
                    const unsigned int* __restrict__ B2p,
                    float* __restrict__ C, float* __restrict__ C2, int M) {
  constexpr int NK = K / 32;
  constexpr int TN = N / 16;
  constexpr int TW = 4;              // N-tiles per wave
  constexpr int NG = TN / TW;        // strip groups
  int lane = threadIdx.x & 31;
  int wid  = (int)((blockIdx.x * blockDim.x + threadIdx.x) >> 5);
  if (wid >= (M >> 4) * NG) return;
  int tm = wid / NG, tg = wid % NG;
  const v8u* ap = (const v8u*)Ap + (size_t)tm * NK * 32 + lane;
  const v8u* bp[TW];
  const v8u* b2p[TW];
#pragma unroll
  for (int t = 0; t < TW; ++t) {
    int tn = tg * TW + t;
    bp[t]  = (const v8u*)Bp  + (size_t)tn * NK * 32 + lane;
    b2p[t] = (const v8u*)B2p + (size_t)tn * NK * 32 + lane;
  }
  v8f acc1[TW], acc2[TW];
#pragma unroll
  for (int t = 0; t < TW; ++t) {
    acc1[t] = (v8f){0.f, 0.f, 0.f, 0.f, 0.f, 0.f, 0.f, 0.f};
    acc2[t] = acc1[t];
  }
#pragma unroll 2
  for (int kk = 0; kk < NK; ++kk) {
    // ---- load phase: all fragments for this K-step in distinct regs ----
    v8u au = ap[(size_t)kk * 32];
    v8u bu[TW], b2u[TW];
#pragma unroll
    for (int t = 0; t < TW; ++t) {
      bu[t]  = bp[t][(size_t)kk * 32];
      b2u[t] = b2p[t][(size_t)kk * 32];
    }
    // ---- compute phase: 2*TW WMMAs sharing one A fragment ----
    v16bf a = __builtin_bit_cast(v16bf, au);
#pragma unroll
    for (int t = 0; t < TW; ++t) {
      acc1[t] = __builtin_amdgcn_wmma_f32_16x16x32_bf16(
          false, a, false, __builtin_bit_cast(v16bf, bu[t]),
          (short)0, acc1[t], false, false);
      acc2[t] = __builtin_amdgcn_wmma_f32_16x16x32_bf16(
          false, a, false, __builtin_bit_cast(v16bf, b2u[t]),
          (short)0, acc2[t], false, false);
    }
  }
  // D layout: VGPR r, lanes 0-15 -> M=r, lanes 16-31 -> M=r+8; N = lane&15
  int mbase = (tm << 4) + ((lane >> 4) << 3);
#pragma unroll
  for (int t = 0; t < TW; ++t) {
    int ncol = ((tg * TW + t) << 4) + (lane & 15);
#pragma unroll
    for (int r = 0; r < 8; ++r) {
      C [(size_t)(mbase + r) * N + ncol] = acc1[t][r];
      C2[(size_t)(mbase + r) * N + ncol] = acc2[t][r];
    }
  }
}

// ---------- per-node attention dots: e_src = H@a_s, e_dst = H@a_d ----------
template<int D>
__global__ __launch_bounds__(256)
void node_dots(const float* __restrict__ H, const float* __restrict__ as,
               const float* __restrict__ ad, float* __restrict__ es,
               float* __restrict__ ed, int n) {
  int wid  = (int)((blockIdx.x * blockDim.x + threadIdx.x) >> 5);
  int lane = threadIdx.x & 31;
  if (wid >= n) return;
  const float* h = H + (size_t)wid * D;
  float s0 = 0.f, s1 = 0.f;
#pragma unroll
  for (int j = lane * 4; j < D; j += 128) {
    float4 hv = *(const float4*)(h + j);
    float4 av = *(const float4*)(as + j);
    float4 dv = *(const float4*)(ad + j);
    s0 += hv.x * av.x + hv.y * av.y + hv.z * av.z + hv.w * av.w;
    s1 += hv.x * dv.x + hv.y * dv.y + hv.z * dv.z + hv.w * dv.w;
  }
  s0 = wave_sum(s0);
  s1 = wave_sum(s1);
  if (lane == 0) { es[wid] = s0; ed[wid] = s1; }
}

// ---------- edge logits + segment max (order-preserving uint key) ----------
__global__ __launch_bounds__(256)
void edge_logits_max(const int* __restrict__ src, const int* __restrict__ dst,
                     const float* __restrict__ w, const float* __restrict__ es,
                     const float* __restrict__ ed, float* __restrict__ logits,
                     unsigned int* __restrict__ mkey, int E) {
  int e = blockIdx.x * blockDim.x + threadIdx.x;
  if (e >= E) return;
  int s = src[e], d = dst[e];
  float x  = es[s] + ed[d];
  float lg = w[e] * (x >= 0.f ? x : 0.2f * x);
  logits[e] = lg;
  unsigned int u   = __builtin_bit_cast(unsigned int, lg);
  unsigned int key = (u & 0x80000000u) ? ~u : (u | 0x80000000u);
  atomicMax(&mkey[s], key);
}

__global__ __launch_bounds__(256)
void decode_m(const unsigned int* __restrict__ mkey, float* __restrict__ m, int n) {
  int i = blockIdx.x * blockDim.x + threadIdx.x;
  if (i >= n) return;
  unsigned int key = mkey[i];
  float f = 0.f;
  if (key != 0u) {
    unsigned int u = (key & 0x80000000u) ? (key & 0x7FFFFFFFu) : ~key;
    f = __builtin_bit_cast(float, u);
    if (!__builtin_isfinite(f)) f = 0.f;
  }
  m[i] = f;
}

// ---------- exp + segment sum + degree (ex stored back into logits buf) ----------
__global__ __launch_bounds__(256)
void edge_exp(const int* __restrict__ src, const float* __restrict__ m,
              float* __restrict__ logits, float* __restrict__ ssum,
              float* __restrict__ deg, int E) {
  int e = blockIdx.x * blockDim.x + threadIdx.x;
  if (e >= E) return;
  int s = src[e];
  float ex = __expf(logits[e] - m[s]);
  logits[e] = ex;
  atomicAdd(&ssum[s], ex);
  atomicAdd(&deg[s], 1.0f);
}

// ---------- message aggregation: Z[src] += alpha * H[dst] ----------
// One thread per (edge, 4-dim group): float4 gather of H, 4 atomics,
// index/alpha loads amortized 4x.
template<int D>
__global__ __launch_bounds__(256)
void aggregate(const int* __restrict__ src, const int* __restrict__ dst,
               const float* __restrict__ ex, const float* __restrict__ ssum,
               const float* __restrict__ H, float* Z, int E) {
  constexpr int GD = D / 4;
  size_t idx   = (size_t)blockIdx.x * blockDim.x + threadIdx.x;
  size_t total = (size_t)E * GD;
  if (idx >= total) return;
  int e  = (int)(idx / GD);
  int j  = (int)(idx & (GD - 1)) * 4;
  int s = src[e], d = dst[e];
  float alpha = ex[e] / (ssum[s] + 1e-16f);
  float4 h = *(const float4*)(H + (size_t)d * D + j);
  float* z = Z + (size_t)s * D + j;
  atomicAdd(z + 0, h.x * alpha);
  atomicAdd(z + 1, h.y * alpha);
  atomicAdd(z + 2, h.z * alpha);
  atomicAdd(z + 3, h.w * alpha);
}

// ---------- LayerNorm(Z+res)*g+b, ELU; isolated nodes use H ----------
template<int D>
__global__ __launch_bounds__(256)
void finalize(const float* Zin, const float* __restrict__ H,
              const float* __restrict__ R, const float* __restrict__ deg,
              const float* __restrict__ g, const float* __restrict__ b,
              float* Zout, int n) {
  constexpr int P4 = D / 128;        // float4 groups per lane
  int wid  = (int)((blockIdx.x * blockDim.x + threadIdx.x) >> 5);
  int lane = threadIdx.x & 31;
  if (wid >= n) return;
  bool iso = deg[wid] <= 0.f;
  float4 x[P4];
  float sum = 0.f;
#pragma unroll
  for (int i = 0; i < P4; ++i) {
    size_t j = (size_t)wid * D + lane * 4 + 128 * i;
    float4 z = iso ? *(const float4*)(H + j) : *(const float4*)(Zin + j);
    float4 r = *(const float4*)(R + j);
    x[i].x = z.x + r.x; x[i].y = z.y + r.y;
    x[i].z = z.z + r.z; x[i].w = z.w + r.w;
    sum += x[i].x + x[i].y + x[i].z + x[i].w;
  }
  float mu = wave_sum(sum) * (1.f / D);
  float vs = 0.f;
#pragma unroll
  for (int i = 0; i < P4; ++i) {
    float dx0 = x[i].x - mu, dx1 = x[i].y - mu, dx2 = x[i].z - mu, dx3 = x[i].w - mu;
    vs += dx0 * dx0 + dx1 * dx1 + dx2 * dx2 + dx3 * dx3;
  }
  float rstd = rsqrtf(wave_sum(vs) * (1.f / D) + 1e-5f);
#pragma unroll
  for (int i = 0; i < P4; ++i) {
    int j = lane * 4 + 128 * i;
    float4 gv = *(const float4*)(g + j);
    float4 bv = *(const float4*)(b + j);
    float4 o;
    o.x = (x[i].x - mu) * rstd * gv.x + bv.x;
    o.y = (x[i].y - mu) * rstd * gv.y + bv.y;
    o.z = (x[i].z - mu) * rstd * gv.z + bv.z;
    o.w = (x[i].w - mu) * rstd * gv.w + bv.w;
    o.x = o.x > 0.f ? o.x : (__expf(o.x) - 1.f);
    o.y = o.y > 0.f ? o.y : (__expf(o.y) - 1.f);
    o.z = o.z > 0.f ? o.z : (__expf(o.z) - 1.f);
    o.w = o.w > 0.f ? o.w : (__expf(o.w) - 1.f);
    *(float4*)(Zout + (size_t)wid * D + j) = o;
  }
}

// ---------- edge reconstruction scores (D2=128: one float4 per lane) ----------
__global__ __launch_bounds__(256)
void edge_scores(const int* __restrict__ ei, const float* __restrict__ Z,
                 float* __restrict__ out, int E) {
  int wid  = (int)((blockIdx.x * blockDim.x + threadIdx.x) >> 5);
  int lane = threadIdx.x & 31;
  if (wid >= E) return;
  int a = ei[wid], c = ei[E + wid];
  float4 za = *(const float4*)(Z + (size_t)a * D2 + lane * 4);
  float4 zb = *(const float4*)(Z + (size_t)c * D2 + lane * 4);
  float s = za.x * zb.x + za.y * zb.y + za.z * zb.z + za.w * zb.w;
  s = wave_sum(s);
  if (lane == 0) out[wid] = 1.f / (1.f + __expf(-s));
}

// ---------- host-side layer driver ----------
template<int Din, int Dout>
static void run_layer(const float* Xin, const int* src, const int* dst, const float* w,
                      const unsigned int* Wp, const unsigned int* Wrp,
                      const float* asv, const float* adv, const float* gv, const float* bv,
                      unsigned int* Ap, float* Hb, float* Rb, float* Zbuf,
                      float* es, float* ed, unsigned int* mkey, float* msh,
                      float* ssum, float* deg, float* elog, hipStream_t stream) {
  int wavesA = (NN / 16) * (Din / 32);
  pack_a<Din><<<(wavesA * 32 + 255) / 256, 256, 0, stream>>>(Xin, Ap, NN);
  int wavesG = (NN / 16) * (Dout / 64);   // TW=4 N-tiles per wave
  gemm_dual_wmma<Din, Dout><<<(wavesG * 32 + 255) / 256, 256, 0, stream>>>(Ap, Wp, Wrp, Hb, Rb, NN);
  node_dots<Dout><<<(NN * 32 + 255) / 256, 256, 0, stream>>>(Hb, asv, adv, es, ed, NN);
  hipMemsetAsync(mkey, 0, (size_t)NN * 4, stream);
  edge_logits_max<<<(EM + 255) / 256, 256, 0, stream>>>(src, dst, w, es, ed, elog, mkey, EM);
  decode_m<<<(NN + 255) / 256, 256, 0, stream>>>(mkey, msh, NN);
  hipMemsetAsync(ssum, 0, (size_t)NN * 4, stream);
  hipMemsetAsync(deg, 0, (size_t)NN * 4, stream);
  edge_exp<<<(EM + 255) / 256, 256, 0, stream>>>(src, msh, elog, ssum, deg, EM);
  hipMemsetAsync(Zbuf, 0, (size_t)NN * Dout * 4, stream);
  size_t totalAgg = (size_t)EM * (Dout / 4);
  aggregate<Dout><<<(unsigned)((totalAgg + 255) / 256), 256, 0, stream>>>(src, dst, elog, ssum, Hb, Zbuf, EM);
  finalize<Dout><<<(NN * 32 + 255) / 256, 256, 0, stream>>>(Zbuf, Hb, Rb, deg, gv, bv, Zbuf, NN);
}

extern "C" void kernel_launch(void* const* d_in, const int* in_sizes, int n_in,
                              void* d_out, int out_size, void* d_ws, size_t ws_size,
                              hipStream_t stream) {
  (void)in_sizes; (void)n_in; (void)out_size; (void)ws_size;
  const float* X   = (const float*)d_in[0];
  const int*   Mei = (const int*)d_in[1];
  const float* Mw  = (const float*)d_in[2];
  const int*   Pei = (const int*)d_in[3];
  const int*   Nei = (const int*)d_in[4];
  const float* W1  = (const float*)d_in[5];
  const float* as1 = (const float*)d_in[6];
  const float* ad1 = (const float*)d_in[7];
  const float* Wr1 = (const float*)d_in[8];
  const float* g1  = (const float*)d_in[9];
  const float* b1  = (const float*)d_in[10];
  const float* W2  = (const float*)d_in[11];
  const float* as2 = (const float*)d_in[12];
  const float* ad2 = (const float*)d_in[13];
  const float* Wr2 = (const float*)d_in[14];
  const float* g2  = (const float*)d_in[15];
  const float* b2  = (const float*)d_in[16];

  float* Zout = (float*)d_out;                 // 20000 x 128
  float* posO = Zout + (size_t)NN * D2;        // 160000
  float* negO = posO + EPOS;                   // 320000

  char* wp = (char*)d_ws;
  auto alloc = [&](size_t bytes) { void* p = wp; wp += (bytes + 255) & ~(size_t)255; return p; };
  unsigned int* Ap   = (unsigned int*)alloc((size_t)NN * D0 * 2);   // packed A frags (reused L2)
  unsigned int* W1p  = (unsigned int*)alloc((size_t)D0 * D1 * 2);
  unsigned int* Wr1p = (unsigned int*)alloc((size_t)D0 * D1 * 2);
  unsigned int* W2p  = (unsigned int*)alloc((size_t)D1 * D2 * 2);
  unsigned int* Wr2p = (unsigned int*)alloc((size_t)D1 * D2 * 2);
  float* Hb   = (float*)alloc((size_t)NN * D1 * 4);
  float* Rb   = (float*)alloc((size_t)NN * D1 * 4);
  float* Z1   = (float*)alloc((size_t)NN * D1 * 4);
  float* es   = (float*)alloc((size_t)NN * 4);
  float* ed   = (float*)alloc((size_t)NN * 4);
  unsigned int* mkey = (unsigned int*)alloc((size_t)NN * 4);
  float* msh  = (float*)alloc((size_t)NN * 4);
  float* ssum = (float*)alloc((size_t)NN * 4);
  float* deg  = (float*)alloc((size_t)NN * 4);
  float* elog = (float*)alloc((size_t)EM * 4);

  const int* src = Mei;
  const int* dst = Mei + EM;

  // pack weights into WMMA B-fragment layout (f32 -> bf16 inline)
  pack_b<D0, D1><<<((D1 / 16) * (D0 / 32) * 32 + 255) / 256, 256, 0, stream>>>(W1,  W1p);
  pack_b<D0, D1><<<((D1 / 16) * (D0 / 32) * 32 + 255) / 256, 256, 0, stream>>>(Wr1, Wr1p);
  pack_b<D1, D2><<<((D2 / 16) * (D1 / 32) * 32 + 255) / 256, 256, 0, stream>>>(W2,  W2p);
  pack_b<D1, D2><<<((D2 / 16) * (D1 / 32) * 32 + 255) / 256, 256, 0, stream>>>(Wr2, Wr2p);

  run_layer<D0, D1>(X,  src, dst, Mw, W1p, Wr1p, as1, ad1, g1, b1,
                    Ap, Hb, Rb, Z1,   es, ed, mkey, msh, ssum, deg, elog, stream);
  run_layer<D1, D2>(Z1, src, dst, Mw, W2p, Wr2p, as2, ad2, g2, b2,
                    Ap, Hb, Rb, Zout, es, ed, mkey, msh, ssum, deg, elog, stream);

  edge_scores<<<(EPOS * 32 + 255) / 256, 256, 0, stream>>>(Pei, Zout, posO, EPOS);
  edge_scores<<<(ENEG * 32 + 255) / 256, 256, 0, stream>>>(Nei, Zout, negO, ENEG);
}
